// GCN_Weights_31533649887980
// MI455X (gfx1250) — matrix operand order repaired and checked by reading, not compiled, not run on hardware
//
#include <hip/hip_runtime.h>
#include <hip/hip_bf16.h>

typedef __attribute__((ext_vector_type(16))) _Float16 v16h;
typedef __attribute__((ext_vector_type(8)))  float    v8f;

#define WMMA_F32_F16(a, b, c) \
  __builtin_amdgcn_wmma_f32_16x16x32_f16(false, (a), false, (b), (short)0, (c), false, false)

// ---------------- small prep kernels ----------------

__global__ void softmax3_kernel(const float* __restrict__ w, float* __restrict__ sp) {
  if (threadIdx.x == 0 && blockIdx.x == 0) {
    float m = fmaxf(w[0], fmaxf(w[1], w[2]));
    float e0 = expf(w[0] - m), e1 = expf(w[1] - m), e2 = expf(w[2] - m);
    float s = e0 + e1 + e2;
    sp[0] = e0 / s; sp[1] = e1 / s; sp[2] = e2 / s;
  }
}

// ew[e]=1 for all edges; deg[n]=1 (self-loop weight folded in here)
__global__ void fill_kernel(float* __restrict__ ew, float* __restrict__ deg, int E, int N) {
  int t = blockIdx.x * blockDim.x + threadIdx.x;
  if (t < E) ew[t] = 1.0f;
  if (t < N) deg[t] = 1.0f;
}

__global__ void scatter_ew_kernel(float* __restrict__ ew,
                                  const int* __restrict__ km, const int* __restrict__ um,
                                  const int* __restrict__ om,
                                  int nk, int nu, int no, const float* __restrict__ sp) {
  int t = blockIdx.x * blockDim.x + threadIdx.x;
  if (t < nk) ew[km[t]] = sp[0];
  if (t < nu) ew[um[t]] = sp[1];
  if (t < no) ew[om[t]] = sp[2];
}

__global__ void deg_accum_kernel(const int* __restrict__ col, const float* __restrict__ ew,
                                 float* __restrict__ deg, int E) {
  int t = blockIdx.x * blockDim.x + threadIdx.x;
  if (t < E) atomicAdd(&deg[col[t]], ew[t]);
}

// deg -> dinv in place
__global__ void dinv_kernel(float* __restrict__ deg, int N) {
  int t = blockIdx.x * blockDim.x + threadIdx.x;
  if (t < N) {
    float d = deg[t];
    deg[t] = (d > 0.0f) ? rsqrtf(d) : 0.0f;
  }
}

// ---------------- WMMA fragment helpers (wave32, contiguous k = (lane/16)*16 + i order) ----------------

// A fragment: 16x32 f16 tile. lane holds row M = lane&15.
// element i -> K = (lane/16)*16 + i
__device__ __forceinline__ int a_frag_k(int i, int lane) {
  return ((lane >> 4) << 4) + i;
}

// B fragment element (lane, e) of tile (n-tile nt, k-chunk kc32):
//   n = nt*16 + (lane&15);  k = kc32*32 + (lane/16)*16 + e
// Bs is stored PRE-PERMUTED: fragment f at Bs[f*512 + lane*16 + e]
// so each lane reads its v16h with one aligned 32-byte LDS access.

// ---------------- embedding GEMM: h = [x[:,0], x[:,1:]@emb_W + emb_b] ----------------
// A = x[:,1:64] (K padded 63->64), B = emb_W padded to 64x64, f32 accumulate.

__global__ __launch_bounds__(256)
void emb_wmma_kernel(const float* __restrict__ x, const float* __restrict__ emb_W,
                     const float* __restrict__ emb_b, _Float16* __restrict__ h, int N) {
  __shared__ __align__(32) _Float16 Bs[8 * 512];       // 8 fragments, lane-packed
  __shared__ __align__(32) _Float16 Stg[8][16 * 64];   // per-wave 16x64 output tile
  int tid = threadIdx.x;
  // fill Bs in fragment-packed order (global reads of tiny emb_W stay L2-hot)
  for (int i = tid; i < 8 * 512; i += 256) {
    int tile = i >> 9, li = i & 511, ln = li >> 4, e = li & 15;
    int nt = tile >> 1, kc = (tile & 1) * 32;
    int n = nt * 16 + (ln & 15);
    int k = kc + ((ln >> 4) << 4) + e;
    Bs[i] = (k < 63 && n < 63) ? (_Float16)emb_W[k * 63 + n] : (_Float16)0.0f;
  }
  __syncthreads();

  int wave = tid >> 5, lane = tid & 31;
  int rowbase = (blockIdx.x * 8 + wave) * 16;
  if (rowbase >= N) return;                 // wave-uniform: EXEC all-ones at WMMA
  _Float16* stg = Stg[wave];

  int hi = lane >> 4;
  int mrow = rowbase + (lane & 15);
  int arow = (mrow < N) ? mrow : (N - 1);   // clamp loads; stores guarded later
  const float* xr = x + (size_t)arow * 64;

  v16h a0, a1;
#pragma unroll
  for (int i = 0; i < 16; ++i) {
    int kk = a_frag_k(i, lane);             // contiguous runs of 8 -> b128 loads
    a0[i] = (_Float16)xr[1 + kk];
    int kk1 = 32 + kk;                      // 63 is zero-pad
    a1[i] = (kk1 < 63) ? (_Float16)xr[1 + kk1] : (_Float16)0.0f;
  }

  // column 0 passthrough into staging (wave-synchronous LDS)
  if (lane < 16) {
    int r = rowbase + lane;
    stg[lane * 64] = (_Float16)x[(size_t)((r < N) ? r : (N - 1)) * 64];
  }

#pragma unroll
  for (int nt = 0; nt < 4; ++nt) {
    const v16h b0 = *(const v16h*)(Bs + (nt * 2 + 0) * 512 + lane * 16);
    const v16h b1 = *(const v16h*)(Bs + (nt * 2 + 1) * 512 + lane * 16);
    v8f c = {};
    c = WMMA_F32_F16(a0, b0, c);
    c = WMMA_F32_F16(a1, b1, c);

    int n = nt * 16 + (lane & 15);
    float bb = (n < 63) ? emb_b[n] : 0.0f;
#pragma unroll
    for (int v = 0; v < 8; ++v) {
      int m = hi * 8 + v;
      if (n < 63) stg[m * 64 + 1 + n] = (_Float16)(c[v] + bb);
    }
  }

  // coalesced writeout: lane -> row lane/2, 64B half-row, 4x b128 stores
  int wr = lane >> 1, wc = (lane & 1) * 32;
  int r = rowbase + wr;
  if (r < N) {
    const uint4* src = (const uint4*)(stg + wr * 64 + wc);
    uint4* dst = (uint4*)(h + (size_t)r * 64 + wc);
#pragma unroll
    for (int q = 0; q < 4; ++q) dst[q] = src[q];
  }
}

// ---------------- GCN GEMM: hW = h(Nx64,f16) @ gcn_W(64x32) ----------------

__global__ __launch_bounds__(256)
void gcn_wmma_kernel(const _Float16* __restrict__ h, const float* __restrict__ gcn_W,
                     float* __restrict__ hW, int N) {
  __shared__ __align__(32) _Float16 Bs[4 * 512];      // 4 fragments, lane-packed
  __shared__ __align__(32) float Stg[8][16 * 32];     // per-wave 16x32 f32 tile
  int tid = threadIdx.x;
  for (int i = tid; i < 4 * 512; i += 256) {
    int tile = i >> 9, li = i & 511, ln = li >> 4, e = li & 15;
    int nt = tile >> 1, kc = (tile & 1) * 32;
    int n = nt * 16 + (ln & 15);
    int k = kc + ((ln >> 4) << 4) + e;
    Bs[i] = (_Float16)gcn_W[k * 32 + n];
  }
  __syncthreads();

  int wave = tid >> 5, lane = tid & 31;
  int rowbase = (blockIdx.x * 8 + wave) * 16;
  if (rowbase >= N) return;
  float* stg = Stg[wave];

  int hi = lane >> 4;
  int mrow = rowbase + (lane & 15);
  int arow = (mrow < N) ? mrow : (N - 1);
  const _Float16* hr = h + (size_t)arow * 64;

  v16h a0, a1;
#pragma unroll
  for (int i = 0; i < 16; ++i) {
    int kk = a_frag_k(i, lane);             // contiguous runs of 8 f16 -> b128
    a0[i] = hr[kk];
    a1[i] = hr[32 + kk];
  }

#pragma unroll
  for (int nt = 0; nt < 2; ++nt) {
    const v16h b0 = *(const v16h*)(Bs + (nt * 2 + 0) * 512 + lane * 16);
    const v16h b1 = *(const v16h*)(Bs + (nt * 2 + 1) * 512 + lane * 16);
    v8f c = {};
    c = WMMA_F32_F16(a0, b0, c);
    c = WMMA_F32_F16(a1, b1, c);

    int n = nt * 16 + (lane & 15);
#pragma unroll
    for (int v = 0; v < 8; ++v) stg[(hi * 8 + v) * 32 + n] = c[v];
  }

  // coalesced writeout: lane -> row lane/2, 16 floats (64B), 4x b128 stores
  int wr = lane >> 1, wc = (lane & 1) * 16;
  int r = rowbase + wr;
  if (r < N) {
    const uint4* src = (const uint4*)(stg + wr * 32 + wc);
    uint4* dst = (uint4*)(hW + (size_t)r * 32 + wc);
#pragma unroll
    for (int q = 0; q < 4; ++q) dst[q] = src[q];
  }
}

// ---------------- aggregation ----------------

// agg = gcn_b + self-loop term dinv^2 * hW   (self-loop norm = dinv*1*dinv)
__global__ void agg_init_kernel(const float* __restrict__ hW, const float* __restrict__ dinv,
                                const float* __restrict__ gcn_b, float* __restrict__ agg, int N) {
  int t = blockIdx.x * blockDim.x + threadIdx.x;
  int n = t >> 5, f = t & 31;
  if (n < N) {
    float di = dinv[n];
    agg[t] = gcn_b[f] + di * di * hW[t];
  }
}

// one wave per edge, lane = feature: coalesced 128B gather of hW[row], 32 contiguous
// global_atomic_add_f32 into agg[col] (both arrays are L2-resident: 12.8MB each)
__global__ void edge_msg_kernel(const int* __restrict__ row, const int* __restrict__ col,
                                const float* __restrict__ ew, const float* __restrict__ dinv,
                                const float* __restrict__ hW, float* __restrict__ agg,
                                long long E) {
  long long t = (long long)blockIdx.x * blockDim.x + threadIdx.x;
  long long e = t >> 5;
  int f = (int)(t & 31);
  if (e >= E) return;
  if (f == 0 && e + 8192 < E) {             // stream-ahead hint -> global_prefetch_b8
    __builtin_prefetch(row + e + 8192, 0, 1);
    __builtin_prefetch(col + e + 8192, 0, 1);
    __builtin_prefetch(ew + e + 8192, 0, 1);
  }
  int r = row[e], c = col[e];
  float nrm = dinv[r] * ew[e] * dinv[c];
  atomicAdd(&agg[(size_t)c * 32 + f], nrm * hW[(size_t)r * 32 + f]);
}

__global__ void zero_kernel(float* __restrict__ p, int n) {
  int t = blockIdx.x * blockDim.x + threadIdx.x;
  if (t < n) p[t] = 0.0f;
}

__global__ void pool_kernel(const float* __restrict__ agg, const int* __restrict__ batch,
                            float* __restrict__ pooled, int N) {
  int t = blockIdx.x * blockDim.x + threadIdx.x;
  int n = t >> 5, f = t & 31;
  if (n < N) {
    float v = fmaxf(agg[t], 0.0f);
    atomicAdd(&pooled[(size_t)batch[n] * 32 + f], v);
  }
}

// tiny MLP head: one thread per graph
__global__ void head_kernel(const float* __restrict__ pooled, const float* __restrict__ fc1_W,
                            const float* __restrict__ fc1_b, const float* __restrict__ out_W,
                            const float* __restrict__ out_b, float* __restrict__ out, int G) {
  int g = blockIdx.x * blockDim.x + threadIdx.x;
  if (g >= G) return;
  const float* p = pooled + (size_t)g * 32;
  float acc = out_b[0];
#pragma unroll 4
  for (int j = 0; j < 32; ++j) {
    float z = fc1_b[j];
#pragma unroll
    for (int i = 0; i < 32; ++i) z += p[i] * fc1_W[i * 32 + j];
    z = fmaxf(z, 0.0f);
    acc += z * out_W[j];
  }
  out[g] = acc;
}

// ---------------- launcher ----------------

extern "C" void kernel_launch(void* const* d_in, const int* in_sizes, int n_in,
                              void* d_out, int out_size, void* d_ws, size_t ws_size,
                              hipStream_t stream) {
  const float* x      = (const float*)d_in[0];
  const int*   eidx   = (const int*)d_in[1];
  const int*   batch  = (const int*)d_in[2];
  const int*   known  = (const int*)d_in[3];
  const int*   unk    = (const int*)d_in[4];
  const int*   obs    = (const int*)d_in[5];
  const float* msg_w  = (const float*)d_in[6];
  const float* emb_W  = (const float*)d_in[7];
  const float* emb_b  = (const float*)d_in[8];
  const float* gcn_W  = (const float*)d_in[9];
  const float* gcn_b  = (const float*)d_in[10];
  const float* fc1_W  = (const float*)d_in[11];
  const float* fc1_b  = (const float*)d_in[12];
  const float* out_W  = (const float*)d_in[13];
  const float* out_b  = (const float*)d_in[14];
  float* out = (float*)d_out;

  const int N  = in_sizes[0] / 64;
  const int E  = in_sizes[1] / 2;
  const int nk = in_sizes[3], nu = in_sizes[4], no = in_sizes[5];
  const int G  = out_size;

  // workspace carve-up (~50 MB total); all segment sizes are 16B multiples
  char* w = (char*)d_ws;
  float*    sp     = (float*)w;     w += 4 * sizeof(float);
  float*    ew     = (float*)w;     w += (size_t)E * sizeof(float);
  float*    dinv   = (float*)w;     w += (size_t)N * sizeof(float);   // deg -> dinv in place
  _Float16* h      = (_Float16*)w;  w += (size_t)N * 64 * sizeof(_Float16);
  float*    hW     = (float*)w;     w += (size_t)N * 32 * sizeof(float);
  float*    agg    = (float*)w;     w += (size_t)N * 32 * sizeof(float);
  float*    pooled = (float*)w;     w += (size_t)G * 32 * sizeof(float);
  (void)ws_size; (void)n_in;

  const int* row = eidx;       // edge_index[0]
  const int* col = eidx + E;   // edge_index[1]
  const int B = 256;

  softmax3_kernel<<<1, 32, 0, stream>>>(msg_w, sp);

  int nfill = (E > N) ? E : N;
  fill_kernel<<<(nfill + B - 1) / B, B, 0, stream>>>(ew, dinv, E, N);

  int nsc = nk; if (nu > nsc) nsc = nu; if (no > nsc) nsc = no;
  scatter_ew_kernel<<<(nsc + B - 1) / B, B, 0, stream>>>(ew, known, unk, obs, nk, nu, no, sp);

  deg_accum_kernel<<<(E + B - 1) / B, B, 0, stream>>>(col, ew, dinv, E);
  dinv_kernel<<<(N + B - 1) / B, B, 0, stream>>>(dinv, N);

  emb_wmma_kernel<<<(N + 127) / 128, B, 0, stream>>>(x, emb_W, emb_b, h, N);
  gcn_wmma_kernel<<<(N + 127) / 128, B, 0, stream>>>(h, gcn_W, hW, N);

  long long nAgg = (long long)N * 32;
  agg_init_kernel<<<(unsigned)((nAgg + B - 1) / B), B, 0, stream>>>(hW, dinv, gcn_b, agg, N);

  long long nEdge = (long long)E * 32;
  edge_msg_kernel<<<(unsigned)((nEdge + B - 1) / B), B, 0, stream>>>(row, col, ew, dinv, hW, agg, E);

  zero_kernel<<<(G * 32 + B - 1) / B, B, 0, stream>>>(pooled, G * 32);
  pool_kernel<<<(unsigned)((nAgg + B - 1) / B), B, 0, stream>>>(agg, batch, pooled, N);

  head_kernel<<<(G + B - 1) / B, B, 0, stream>>>(pooled, fc1_W, fc1_b, out_W, out_b, out, G);
}
